// Quantizer_55989193671194
// MI455X (gfx1250) — matrix operand order repaired and checked by reading, not compiled, not run on hardware
//
#include <hip/hip_runtime.h>
#include <cstdint>
#include <cstddef>

// ---------------- problem constants ----------------
#define R_STEPS 8
#define G_GRP   2
#define K_CODES 1024
#define E_DIM   256
#define B_SZ    16
#define C_SZ    512
#define T_SZ    2048
#define N_TOK   (B_SZ*T_SZ)          // 32768 tokens
#define M_TILE  128                   // tokens per workgroup
#define NTHREADS 256                  // 8 waves (wave32)

// ---------------- LDS layout (bytes) ----------------
#define XRES_STRIDE 257                               // floats per row (pad 1 -> conflict-free)
#define XRES_OFF    0
#define XRES_BYTES  (M_TILE*XRES_STRIDE*4)            // 131584
#define A_STRIDE    264                               // halfs per row (pad 8 -> 16B aligned rows)
#define A_OFF       (XRES_OFF + XRES_BYTES)           // 131584
#define A_BYTES     (M_TILE*A_STRIDE*2)               // 67584
#define CB_ROWB     528                               // bytes per code row (264 halfs)
#define CB_OFF      (A_OFF + A_BYTES)                 // 199168
#define CB_BUFB     (64*CB_ROWB)                      // 33792 (one 64-code chunk)
#define CB_BYTES    (2*CB_BUFB)                       // double buffered
#define CN_OFF      (CB_OFF + CB_BYTES)               // 266752
#define CN_BYTES    (K_CODES*4)                       // 4096
#define IDX_OFF     (CN_OFF + CN_BYTES)               // 270848
#define IDX_BYTES   (M_TILE*4)                        // 512
#define RED_OFF     (IDX_OFF + IDX_BYTES)             // 271360
#define LDS_BYTES   (RED_OFF + 32)                    // 271392

typedef _Float16 v16h __attribute__((ext_vector_type(16)));
typedef _Float16 v8h  __attribute__((ext_vector_type(8)));
typedef float    v8f  __attribute__((ext_vector_type(8)));

union HPair { struct { v8h lo; v8h hi; } p; v16h v; };

// Issue one 64-code f16 codebook chunk: global -> LDS, async (ASYNCcnt).
// 2048 x b128 transfers spread over 256 threads x 8 issues, 16B per lane each.
__device__ __forceinline__ void issue_chunk(const _Float16* cbStep, int chunk,
                                            int buf, int tid) {
  const uint64_t sbase =
      (uint64_t)(uintptr_t)cbStep + (uint64_t)chunk * (64u * E_DIM * 2u);
#pragma unroll
  for (int j = 0; j < 8; ++j) {
    const int item = tid + j * 256;
    const int row  = item >> 5;         // code row within chunk (32 x b128 per row)
    const int col  = item & 31;
    const uint32_t voff = (uint32_t)((row * E_DIM + col * 8) * 2);   // bytes in global
    const uint32_t ldst = (uint32_t)(CB_OFF + buf * CB_BUFB + row * CB_ROWB + col * 16);
    asm volatile("global_load_async_to_lds_b128 %0, %1, %2 offset:0"
                 :: "v"(ldst), "v"(voff), "s"(sbase) : "memory");
  }
}

__global__ __launch_bounds__(NTHREADS) void rvq_main(
    const float* __restrict__ xin, const float* __restrict__ cb32,
    const _Float16* __restrict__ cb16, const float* __restrict__ cnormG,
    float* __restrict__ qout, float* __restrict__ lossP,
    float* __restrict__ idxO) {
  extern __shared__ char smem[];
  float*    xres = (float*)(smem + XRES_OFF);
  _Float16* a16  = (_Float16*)(smem + A_OFF);
  _Float16* cbb  = (_Float16*)(smem + CB_OFF);
  float*    cns  = (float*)(smem + CN_OFF);
  int*      idxs = (int*)(smem + IDX_OFF);
  float*    red  = (float*)(smem + RED_OFF);

  const int tid     = threadIdx.x;
  const int lane    = tid & 31;
  const int wv      = tid >> 5;          // wave 0..7
  const int laneN   = lane & 15;
  const int laneHi  = lane >> 4;         // 0: lanes 0-15, 1: lanes 16-31
  const int rowBase = wv * 16;           // this wave's 16 token rows

  const int g     = blockIdx.x & 1;
  const int tile  = (int)blockIdx.x >> 1;
  const int n0    = tile * M_TILE;       // first token (n = b*T + t)
  const int bIdx  = n0 >> 11;            // / T_SZ
  const int tbase = n0 & (T_SZ - 1);

  // ---- load xin tile -> xres (fp32).  x[n, e] = xin[b, g*256+e, t] ----
  {
    const int m = tid & 127;             // token within tile (coalesced over t)
    const int esel = tid >> 7;           // 0/1
#pragma unroll 4
    for (int it = 0; it < 128; ++it) {
      const int e = it * 2 + esel;
      xres[m * XRES_STRIDE + e] =
          xin[(size_t)(bIdx * C_SZ + g * E_DIM + e) * T_SZ + tbase + m];
    }
  }

  float lossAcc = 0.f;

  for (int r = 0; r < R_STEPS; ++r) {
    __syncthreads();                     // residual tile coherent across waves

    // ---- rebuild f16 A tile (WMMA A-friendly row-major, padded rows) ----
    {
      const int m = tid >> 1;
      const float* src = xres + m * XRES_STRIDE + (tid & 1) * 128;
      _Float16*    dst = a16 + m * A_STRIDE + (tid & 1) * 128;
#pragma unroll 8
      for (int e = 0; e < 128; ++e) dst[e] = (_Float16)src[e];
    }
    // ---- preload 0.5*||c||^2 for (r,g) ----
    {
      const float* cg = cnormG + (r * G_GRP + g) * K_CODES;
#pragma unroll
      for (int j = 0; j < 4; ++j) cns[tid + j * 256] = cg[tid + j * 256];
    }

    const _Float16* cbStep =
        cb16 + (size_t)(r * G_GRP + g) * K_CODES * E_DIM;
    issue_chunk(cbStep, 0, 0, tid);      // prefetch first chunk
    __syncthreads();                     // A tile + cnorms ready

    // ---- cache this wave's A rows in registers (8 K-chunks of 32) ----
    v16h aR[8];
    {
      const _Float16* abase = a16 + (rowBase + laneN) * A_STRIDE + laneHi * 8;
#pragma unroll
      for (int kc = 0; kc < 8; ++kc) {
        HPair u;
        u.p.lo = *(const v8h*)(abase + kc * 32);        // K +0..7   (or 8..15)
        u.p.hi = *(const v8h*)(abase + kc * 32 + 16);   // K +16..23 (or 24..31)
        aR[kc] = u.v;
      }
    }

    float best[8];
    int   bestk[8];
#pragma unroll
    for (int i = 0; i < 8; ++i) { best[i] = -3.4e38f; bestk[i] = 0; }

    // ---- sweep K=1024 codes in 16 double-buffered chunks of 64 ----
    for (int chunk = 0; chunk < 16; ++chunk) {
      asm volatile("s_wait_asynccnt 0" ::: "memory");   // my chunk data landed
      __syncthreads();                                  // everyone's landed; prev compute done
      if (chunk < 15) issue_chunk(cbStep, chunk + 1, (chunk + 1) & 1, tid);

      const _Float16* cbuf = cbb + (chunk & 1) * (CB_BUFB / 2);
#pragma unroll
      for (int p = 0; p < 2; ++p) {                     // 2 concurrent 16-code tiles
        v8f acc0 = {}; v8f acc1 = {};
        // B layout: lanes 0-15 hold K 0..7,16..23, lanes 16-31 hold K 8..15,24..31 per 32-K block;
        // column n = code vector (row-major in LDS) -> two 16B pieces per lane.
        const _Float16* b0 = cbuf + (p * 32 + laneN) * A_STRIDE + laneHi * 8;
        const _Float16* b1 = b0 + 16 * A_STRIDE;
#pragma unroll
        for (int kc = 0; kc < 8; ++kc) {
          HPair u0, u1;
          u0.p.lo = *(const v8h*)(b0 + kc * 32);
          u0.p.hi = *(const v8h*)(b0 + kc * 32 + 16);
          u1.p.lo = *(const v8h*)(b1 + kc * 32);
          u1.p.hi = *(const v8h*)(b1 + kc * 32 + 16);
          acc0 = __builtin_amdgcn_wmma_f32_16x16x32_f16(
              false, aR[kc], false, u0.v, (short)0, acc0, false, false);
          acc1 = __builtin_amdgcn_wmma_f32_16x16x32_f16(
              false, aR[kc], false, u1.v, (short)0, acc1, false, false);
        }
        const int   k0  = chunk * 64 + p * 32 + laneN;
        const int   k1  = k0 + 16;
        const float cn0 = cns[k0];
        const float cn1 = cns[k1];
#pragma unroll
        for (int i = 0; i < 8; ++i) {                   // score = x.c - 0.5||c||^2
          const float s0 = acc0[i] - cn0;
          if (s0 > best[i]) { best[i] = s0; bestk[i] = k0; }
          const float s1 = acc1[i] - cn1;
          if (s1 > best[i]) { best[i] = s1; bestk[i] = k1; }
        }
      }
    }

    // ---- argmax across the 16 lanes of each half (C layout: N = lane%16) ----
#pragma unroll
    for (int off = 8; off >= 1; off >>= 1) {
#pragma unroll
      for (int i = 0; i < 8; ++i) {
        const float ov = __shfl_xor(best[i], off, 32);
        const int   ok = __shfl_xor(bestk[i], off, 32);
        if (ov > best[i] || (ov == best[i] && ok < bestk[i])) {
          best[i] = ov; bestk[i] = ok;
        }
      }
    }

    // acc VGPR i: lanes 0-15 -> row i, lanes 16-31 -> row 8+i
    if (laneN == 0) {
#pragma unroll
      for (int i = 0; i < 8; ++i) {
        const int row = rowBase + laneHi * 8 + i;
        idxs[row] = bestk[i];
        idxO[(size_t)(r * G_GRP + g) * N_TOK + n0 + row] = (float)bestk[i];
      }
    }

    // ---- residual update in fp32 (own wave's 16 rows) + loss accumulation ----
    const float* cbg = cb32 + (size_t)(r * G_GRP + g) * K_CODES * E_DIM;
    for (int i = 0; i < 16; ++i) {
      const int row = rowBase + i;
      const int k   = idxs[row];
      const float* cw = cbg + (size_t)k * E_DIM;
      float* xr = xres + row * XRES_STRIDE;
#pragma unroll
      for (int j = 0; j < 8; ++j) {
        const int e = lane + 32 * j;
        const float nr = xr[e] - cw[e];   // new residual; (z_q - x) = -nr
        xr[e] = nr;
        lossAcc += nr * nr;
      }
    }
  }

  __syncthreads();
  // ---- quantized_out = xin - residual_final ----
  {
    const int m = tid & 127;
    const int esel = tid >> 7;
#pragma unroll 4
    for (int it = 0; it < 128; ++it) {
      const int e = it * 2 + esel;
      const size_t gi = (size_t)(bIdx * C_SZ + g * E_DIM + e) * T_SZ + tbase + m;
      qout[gi] = xin[gi] - xres[m * XRES_STRIDE + e];
    }
  }

  // ---- loss reduction: lanes -> waves -> one atomicAdd ----
#pragma unroll
  for (int off = 16; off >= 1; off >>= 1) lossAcc += __shfl_xor(lossAcc, off, 32);
  if (lane == 0) red[wv] = lossAcc;
  __syncthreads();
  if (tid == 0) {
    float s = 0.f;
#pragma unroll
    for (int w = 0; w < 8; ++w) s += red[w];
    // (CL+CM) / (R * B*T*C) = 1.25 / 134217728
    atomicAdd(lossP, s * (1.25f / 134217728.0f));
  }
}

// ---------------- prep kernels ----------------
__global__ __launch_bounds__(256) void prep_convert(
    const float* __restrict__ cb, _Float16* __restrict__ cb16,
    float* __restrict__ lossP) {
  const size_t base = (size_t)blockIdx.x * 1024 + threadIdx.x;
#pragma unroll
  for (int j = 0; j < 4; ++j) cb16[base + j * 256] = (_Float16)cb[base + j * 256];
  if (blockIdx.x == 0 && threadIdx.x == 0) *lossP = 0.f;   // zero loss slot each launch
}

__global__ __launch_bounds__(256) void prep_cnorm(
    const float* __restrict__ cb, float* __restrict__ cn) {
  const int lane = threadIdx.x & 31, wv = threadIdx.x >> 5;
  const int row  = blockIdx.x * 8 + wv;      // one code vector per wave
  const float* p = cb + (size_t)row * E_DIM;
  float s = 0.f;
#pragma unroll
  for (int j = 0; j < 8; ++j) { const float v = p[lane + 32 * j]; s += v * v; }
#pragma unroll
  for (int off = 16; off >= 1; off >>= 1) s += __shfl_xor(s, off, 32);
  if (lane == 0) cn[row] = 0.5f * s;
}

// ---------------- host launcher ----------------
extern "C" void kernel_launch(void* const* d_in, const int* in_sizes, int n_in,
                              void* d_out, int out_size, void* d_ws, size_t ws_size,
                              hipStream_t stream) {
  (void)in_sizes; (void)n_in; (void)out_size; (void)ws_size;
  const float* xin = (const float*)d_in[0];
  const float* cb  = (const float*)d_in[1];

  _Float16* cb16 = (_Float16*)d_ws;  // 8.39 MB f16 codebooks
  float* cnorm   = (float*)((char*)d_ws + (size_t)R_STEPS * G_GRP * K_CODES * E_DIM * 2);
  float* qout    = (float*)d_out;
  float* lossP   = qout + (size_t)B_SZ * C_SZ * T_SZ;   // offset 16777216
  float* idxO    = lossP + 1;                           // [R, G, N] as float

  prep_convert<<<4096, 256, 0, stream>>>(cb, cb16, lossP);
  prep_cnorm<<<(R_STEPS * G_GRP * K_CODES) / 8, 256, 0, stream>>>(cb, cnorm);

  (void)hipFuncSetAttribute((const void*)rvq_main,
                            hipFuncAttributeMaxDynamicSharedMemorySize, LDS_BYTES);
  rvq_main<<<(N_TOK / M_TILE) * G_GRP, NTHREADS, LDS_BYTES, stream>>>(
      xin, cb, cb16, cnorm, qout, lossP, idxO);
}
